// GraphRNNDecoder_60155311947970
// MI455X (gfx1250) — matrix operand order repaired and checked by reading, not compiled, not run on hardware
//
#include <hip/hip_runtime.h>
#include <math.h>

// ---- model constants (match reference) ----
#define BB      16
#define TT      16
#define NNODES  30
#define DINP    8
#define HH      256
#define ETYPES  4
#define NHEADS  8
#define DH      32
#define LLAYERS 2
#define FFND    1024
#define EDGES   (NNODES * (NNODES - 1))   // 870
#define NTOK    (BB * NNODES)             // 480
#define LNEPS   1e-5f
#define ASCALE  0.17677669529663687f      // 1/sqrt(32)
#define EDGE_NORM_INV (1.0f / 3.0f)       // 1/(ET-1)

typedef __bf16 v16bf __attribute__((ext_vector_type(16)));
typedef __bf16 v8bf  __attribute__((ext_vector_type(8)));
typedef float  v8f   __attribute__((ext_vector_type(8)));

__device__ __forceinline__ __bf16 f2bf(float f) {
  union { float f; unsigned u; } x; x.f = f;
  unsigned r = x.u + 0x7FFFu + ((x.u >> 16) & 1u);   // round-to-nearest-even
  unsigned short h = (unsigned short)(r >> 16);
  __bf16 b; __builtin_memcpy(&b, &h, 2); return b;
}
__device__ __forceinline__ float bf2f(__bf16 b) {
  unsigned short h; __builtin_memcpy(&h, &b, 2);
  union { unsigned u; float f; } x; x.u = ((unsigned)h) << 16;
  return x.f;
}

enum { A_NORMAL = 0, A_GATHER = 1 };
enum { E_NONE = 0, E_RELU = 1, E_TANH = 2, E_TANHEDGE = 3, E_ADDRES = 4 };

// ---------------------------------------------------------------------------
// bf16 WMMA GEMM: C(M x Nout) = epi( A(M x K) @ W^T + bias ),  W^T given as
// Wt (K x Np) bf16, Np = ceil32(Nout), zero-padded.  Requirements (enforced
// by the caller): M%16==0, K%32==0, Np%32==0.  One wave computes a 16x32 C
// tile (2 WMMA subtiles sharing the A fragment).  No lane-divergent control
// flow in the K loop: 2 b128 A loads, 32 16-bit B loads, 2 v_wmma per iteration.
// AMODE==A_GATHER: A rows gathered on the fly from hidden (bf16) via the
// deterministic SEND/RECV enumeration of the all-ones-minus-eye adjacency.
// EPI==E_ADDRES writes fp32 (pred head); everything else writes bf16.
// ---------------------------------------------------------------------------
template<int AMODE, int EPI>
__global__ __launch_bounds__(256) void wmma_gemm(
    const __bf16* __restrict__ A, const __bf16* __restrict__ hid,
    const __bf16* __restrict__ Wt, const float* __restrict__ bias,
    const float* __restrict__ extra, void* __restrict__ Cout,
    int M, int Nout, int Np, int K, int tilesN2, int et, int initacc)
{
  int wid  = threadIdx.x >> 5;
  int lane = threadIdx.x & 31;
  int tile = blockIdx.x * 8 + wid;
  int tilesM = M >> 4;
  if (tile >= tilesM * tilesN2) return;        // wave-uniform exit
  int tm = tile / tilesN2, tn2 = tile - tm * tilesN2;
  int half = lane >> 4, lrow = lane & 15;

  int arow = tm * 16 + lrow;                   // A-fragment row for this lane
  const __bf16* hidR = nullptr;
  const __bf16* hidS = nullptr;
  if (AMODE == A_GATHER) {
    int gb = arow / EDGES;
    int e  = arow - gb * EDGES;
    int gsend = e / (NNODES - 1);              // SEND[e]
    int wdx = e - gsend * (NNODES - 1);
    int grecv = wdx + (wdx >= gsend ? 1 : 0);  // RECV[e]
    hidR = hid + (size_t)(gb * NNODES + grecv) * HH;
    hidS = hid + (size_t)(gb * NNODES + gsend) * HH;
  }
  const __bf16* Arow = A + (size_t)arow * K;

  v8f acc0 = {0.f,0.f,0.f,0.f,0.f,0.f,0.f,0.f};
  v8f acc1 = acc0;

  for (int k0 = 0; k0 < K; k0 += 32) {
    v8bf a0, a1;
    if (AMODE == A_GATHER) {
      int kr0 = k0 + half * 8;                 // 8-elem run, never straddles HH
      int kr1 = kr0 + 16;
      const __bf16* p0 = (kr0 < HH) ? (hidR + kr0) : (hidS + (kr0 - HH));
      const __bf16* p1 = (kr1 < HH) ? (hidR + kr1) : (hidS + (kr1 - HH));
      a0 = *(const v8bf*)p0;
      a1 = *(const v8bf*)p1;
    } else {
      a0 = *(const v8bf*)(Arow + k0 + half * 8);
      a1 = *(const v8bf*)(Arow + k0 + 16 + half * 8);
    }
    v16bf af = __builtin_shufflevector(a0, a1, 0,1,2,3,4,5,6,7,8,9,10,11,12,13,14,15);

    const __bf16* bp = Wt + (size_t)k0 * Np + tn2 * 32 + lrow;
    v16bf bfrag0, bfrag1;
#pragma unroll
    for (int s = 0; s < 16; ++s) {
      int kk = (s < 8) ? (half * 8 + s) : (16 + half * 8 + (s - 8));
      bfrag0[s] = bp[(size_t)kk * Np];
      bfrag1[s] = bp[(size_t)kk * Np + 16];
    }

    acc0 = __builtin_amdgcn_wmma_f32_16x16x32_bf16(false, af, false, bfrag0, (short)0, acc0, false, false);
    acc1 = __builtin_amdgcn_wmma_f32_16x16x32_bf16(false, af, false, bfrag1, (short)0, acc1, false, false);
  }

  // C layout: vgpr r -> M = 8*half + r, N = lane%16
#pragma unroll
  for (int sub = 0; sub < 2; ++sub) {
    int nn = tn2 * 32 + sub * 16 + lrow;
    if (nn < Nout) {
#pragma unroll
      for (int r = 0; r < 8; ++r) {
        int m = tm * 16 + (half << 3) + r;
        float v = (sub == 0) ? acc0[r] : acc1[r];
        if (bias) v += bias[nn];
        if (EPI == E_RELU)      v = fmaxf(v, 0.f);
        else if (EPI == E_TANH) v = tanhf(v);
        else if (EPI == E_TANHEDGE) {
          int b = m / EDGES; int e = m - b * EDGES;
          float w = extra[((size_t)b * EDGES + e) * ETYPES + et] * EDGE_NORM_INV;
          v = tanhf(v) * w;
          if (!initacc) v += bf2f(((const __bf16*)Cout)[(size_t)m * Nout + nn]);
        } else if (EPI == E_ADDRES) {
          v += extra[(size_t)m * Nout + nn];
        }
        if (EPI == E_ADDRES) ((float*)Cout)[(size_t)m * Nout + nn] = v;
        else                 ((__bf16*)Cout)[(size_t)m * Nout + nn] = f2bf(v);
      }
    }
  }
}

// Weight convert + transpose: W (Nout x K) fp32 -> Wt (Kpad x Np) bf16,
// zero-padded.  Np = ceil32(Nout).
__global__ void wtrans_kernel(const float* __restrict__ W, __bf16* __restrict__ Wt,
                              int Nout, int K, int Kpad, int Np) {
  int idx = blockIdx.x * 256 + threadIdx.x;
  if (idx >= Kpad * Np) return;
  int k = idx / Np, n = idx - k * Np;
  float v = (k < K && n < Nout) ? W[(size_t)n * K + k] : 0.f;
  Wt[idx] = f2bf(v);
}

// ---------------------------------------------------------------------------
// Softmax attention (bf16 in/out), one (batch, head) per block, one query per
// thread.  Seq len = NNODES = 30, dh = 32.  Output row stride = HH.
// ---------------------------------------------------------------------------
__global__ __launch_bounds__(32) void attn_kernel(
    const __bf16* __restrict__ qb, int ldq,
    const __bf16* __restrict__ kb, int ldk,
    const __bf16* __restrict__ vb, int ldv,
    __bf16* __restrict__ outb)
{
  int bh = blockIdx.x;
  int b = bh / NHEADS, h = bh - b * NHEADS;
  int i = threadIdx.x;
  if (i >= NNODES) return;
  int co = h * DH;

  float q[DH];
#pragma unroll
  for (int d = 0; d < DH; ++d)
    q[d] = bf2f(qb[(size_t)(b * NNODES + i) * ldq + co + d]) * ASCALE;

  float s[NNODES];
  float mx = -1e30f;
#pragma unroll
  for (int j = 0; j < NNODES; ++j) {
    float dot = 0.f;
#pragma unroll
    for (int d = 0; d < DH; ++d)
      dot += q[d] * bf2f(kb[(size_t)(b * NNODES + j) * ldk + co + d]);
    s[j] = dot; mx = fmaxf(mx, dot);
  }
  float sum = 0.f;
#pragma unroll
  for (int j = 0; j < NNODES; ++j) { s[j] = expf(s[j] - mx); sum += s[j]; }

  float acc[DH];
#pragma unroll
  for (int d = 0; d < DH; ++d) acc[d] = 0.f;
#pragma unroll
  for (int j = 0; j < NNODES; ++j) {
    float w = s[j];
#pragma unroll
    for (int d = 0; d < DH; ++d)
      acc[d] += w * bf2f(vb[(size_t)(b * NNODES + j) * ldv + co + d]);
  }
  float inv = 1.f / sum;
#pragma unroll
  for (int d = 0; d < DH; ++d)
    outb[(size_t)(b * NNODES + i) * HH + co + d] = f2bf(acc[d] * inv);
}

// LayerNorm over D=HH (bf16 in/out, fp32 math), one row per block.
__global__ __launch_bounds__(256) void ln_kernel(
    const __bf16* __restrict__ x, const __bf16* __restrict__ res,
    const float* __restrict__ g, const float* __restrict__ b,
    __bf16* __restrict__ out)
{
  __shared__ float sm[HH];
  int row = blockIdx.x, t = threadIdx.x;
  float v = bf2f(x[(size_t)row * HH + t]);
  if (res) v += bf2f(res[(size_t)row * HH + t]);
  sm[t] = v; __syncthreads();
  for (int s = 128; s > 0; s >>= 1) { if (t < s) sm[t] += sm[t + s]; __syncthreads(); }
  float mu = sm[0] * (1.0f / HH);
  __syncthreads();
  float d = v - mu;
  sm[t] = d * d; __syncthreads();
  for (int s = 128; s > 0; s >>= 1) { if (t < s) sm[t] += sm[t + s]; __syncthreads(); }
  float var = sm[0] * (1.0f / HH);
  out[(size_t)row * HH + t] = f2bf(d * rsqrtf(var + LNEPS) * g[t] + b[t]);
}

// agg[b,c,:] = sum over senders r!=c of msgs[b, e(r,c), :] / (N-1)
__global__ __launch_bounds__(256) void agg_kernel(
    const __bf16* __restrict__ msgs, __bf16* __restrict__ agg)
{
  int idx = blockIdx.x * 256 + threadIdx.x;          // over B*NNODES*HH
  if (idx >= BB * NNODES * HH) return;
  int hcol = idx & (HH - 1);
  int bn = idx >> 8;                                  // HH == 256
  int c = bn % NNODES, b = bn / NNODES;
  float s = 0.f;
#pragma unroll
  for (int r = 0; r < NNODES; ++r) {
    if (r == c) continue;
    int e = r * (NNODES - 1) + c - (c > r ? 1 : 0);
    s += bf2f(msgs[((size_t)b * EDGES + e) * HH + hcol]);
  }
  agg[idx] = f2bf(s * (1.0f / (NNODES - 1)));
}

__global__ void zero_bf_kernel(__bf16* __restrict__ p, int n) {
  int i = blockIdx.x * 256 + threadIdx.x;
  if (i < n) p[i] = f2bf(0.f);
}

// ins = inputs[:, 0]   (inputs is (B,T,N,DIN), fp32 state)
__global__ void init_ins_kernel(const float* __restrict__ inputs, float* __restrict__ ins) {
  int i = blockIdx.x * 256 + threadIdx.x;
  if (i >= BB * NNODES * DINP) return;
  int d = i % DINP; int rest = i / DINP;
  int n = rest % NNODES; int b = rest / NNODES;
  ins[i] = inputs[(((size_t)b * TT + 0) * NNODES + n) * DINP + d];
}

// fp32 ins (NTOK x 8) -> bf16 padded (NTOK x 32) for the K%32==0 GEMM rule
__global__ void pack_ins_kernel(const float* __restrict__ ins, __bf16* __restrict__ insb) {
  int i = blockIdx.x * 256 + threadIdx.x;             // over NTOK*32
  if (i >= NTOK * 32) return;
  int d = i & 31; int row = i >> 5;
  insb[i] = f2bf(d < DINP ? ins[row * DINP + d] : 0.f);
}

// out[:, t] = pred   (out is (B,T,N,DIN))
__global__ void write_out_kernel(const float* __restrict__ pred, float* __restrict__ out, int t) {
  int i = blockIdx.x * 256 + threadIdx.x;
  if (i >= BB * NNODES * DINP) return;
  int d = i % DINP; int rest = i / DINP;
  int n = rest % NNODES; int b = rest / NNODES;
  out[(((size_t)b * TT + t) * NNODES + n) * DINP + d] = pred[i];
}

// ---------------------------------------------------------------------------
// Host-side dispatch
// ---------------------------------------------------------------------------
static inline int ceil32(int n) { return (n + 31) & ~31; }

static void gemm(hipStream_t st, int amode, int epi,
                 const __bf16* A, const __bf16* hid, const __bf16* Wt,
                 const float* bias, const float* extra, void* C,
                 int M, int Nout, int K, int et = 0, int initacc = 0)
{
  int Np = ceil32(Nout);
  int tilesM = M / 16, tilesN2 = Np / 32;
  dim3 grid((tilesM * tilesN2 + 7) / 8), blk(256);
  if (amode == A_GATHER) {
    wmma_gemm<A_GATHER, E_TANH><<<grid, blk, 0, st>>>(A, hid, Wt, bias, extra, C, M, Nout, Np, K, tilesN2, et, initacc);
    return;
  }
  switch (epi) {
    case E_NONE:     wmma_gemm<A_NORMAL, E_NONE    ><<<grid, blk, 0, st>>>(A, hid, Wt, bias, extra, C, M, Nout, Np, K, tilesN2, et, initacc); break;
    case E_RELU:     wmma_gemm<A_NORMAL, E_RELU    ><<<grid, blk, 0, st>>>(A, hid, Wt, bias, extra, C, M, Nout, Np, K, tilesN2, et, initacc); break;
    case E_TANH:     wmma_gemm<A_NORMAL, E_TANH    ><<<grid, blk, 0, st>>>(A, hid, Wt, bias, extra, C, M, Nout, Np, K, tilesN2, et, initacc); break;
    case E_TANHEDGE: wmma_gemm<A_NORMAL, E_TANHEDGE><<<grid, blk, 0, st>>>(A, hid, Wt, bias, extra, C, M, Nout, Np, K, tilesN2, et, initacc); break;
    default:         wmma_gemm<A_NORMAL, E_ADDRES  ><<<grid, blk, 0, st>>>(A, hid, Wt, bias, extra, C, M, Nout, Np, K, tilesN2, et, initacc); break;
  }
}

static void wtrans(hipStream_t st, const float* W, __bf16* Wt, int Nout, int K, int Kpad) {
  int Np = ceil32(Nout);
  int n = Kpad * Np;
  wtrans_kernel<<<(n + 255) / 256, 256, 0, st>>>(W, Wt, Nout, K, Kpad, Np);
}

extern "C" void kernel_launch(void* const* d_in, const int* in_sizes, int n_in,
                              void* d_out, int out_size, void* d_ws, size_t ws_size,
                              hipStream_t stream) {
  (void)in_sizes; (void)n_in; (void)out_size; (void)ws_size;
  // ---- inputs, in setup_inputs() flattened dict order ----
  const float* inputs   = (const float*)d_in[0];   // (B,T,N,DIN)
  const float* sedges   = (const float*)d_in[1];   // (B,E,ET)
  const float* fc1_w    = (const float*)d_in[2];   // (ET,H,2H)
  const float* fc1_b    = (const float*)d_in[3];   // (ET,H)
  const float* fc2_w    = (const float*)d_in[4];   // (ET,H,H)
  const float* fc2_b    = (const float*)d_in[5];   // (ET,H)
  const float* in_w     = (const float*)d_in[6];   // (H,DIN)
  const float* in_b     = (const float*)d_in[7];
  const float* o1_w     = (const float*)d_in[8];
  const float* o1_b     = (const float*)d_in[9];
  const float* o2_w     = (const float*)d_in[10];
  const float* o2_b     = (const float*)d_in[11];
  const float* o3_w     = (const float*)d_in[12];  // (DIN,H)
  const float* o3_b     = (const float*)d_in[13];
  const float* e_qkv_w  = (const float*)d_in[14];  // (L,3H,H)
  const float* e_qkv_b  = (const float*)d_in[15];
  const float* e_o_w    = (const float*)d_in[16];
  const float* e_o_b    = (const float*)d_in[17];
  const float* e_ff1_w  = (const float*)d_in[18];  // (L,FFN,H)
  const float* e_ff1_b  = (const float*)d_in[19];
  const float* e_ff2_w  = (const float*)d_in[20];  // (L,H,FFN)
  const float* e_ff2_b  = (const float*)d_in[21];
  const float* e_ln1_g  = (const float*)d_in[22];
  const float* e_ln1_b  = (const float*)d_in[23];
  const float* e_ln2_g  = (const float*)d_in[24];
  const float* e_ln2_b  = (const float*)d_in[25];
  const float* e_nrm_g  = (const float*)d_in[26];
  const float* e_nrm_b  = (const float*)d_in[27];
  const float* d_qkv_w  = (const float*)d_in[28];
  const float* d_qkv_b  = (const float*)d_in[29];
  const float* d_o_w    = (const float*)d_in[30];
  const float* d_o_b    = (const float*)d_in[31];
  const float* d_xqkv_w = (const float*)d_in[32];
  const float* d_xqkv_b = (const float*)d_in[33];
  const float* d_xo_w   = (const float*)d_in[34];
  const float* d_xo_b   = (const float*)d_in[35];
  const float* d_ff1_w  = (const float*)d_in[36];
  const float* d_ff1_b  = (const float*)d_in[37];
  const float* d_ff2_w  = (const float*)d_in[38];
  const float* d_ff2_b  = (const float*)d_in[39];
  const float* d_ln1_g  = (const float*)d_in[40];
  const float* d_ln1_b  = (const float*)d_in[41];
  const float* d_ln2_g  = (const float*)d_in[42];
  const float* d_ln2_b  = (const float*)d_in[43];
  const float* d_ln3_g  = (const float*)d_in[44];
  const float* d_ln3_b  = (const float*)d_in[45];
  const float* d_nrm_g  = (const float*)d_in[46];
  const float* d_nrm_b  = (const float*)d_in[47];
  float* out = (float*)d_out;

  // ---- workspace carve ----
  char* wsb = (char*)d_ws;
  size_t off = 0;
  auto allocf = [&](size_t n) { float* p = (float*)(wsb + off); off += n * 4; return p; };
  auto allocb = [&](size_t n) { __bf16* p = (__bf16*)(wsb + off); off += ((n + 15) & ~(size_t)15) * 2; return p; };

  // fp32 state
  float* ins0 = allocf((size_t)BB * NNODES * DINP);
  float* ins1 = allocf((size_t)BB * NNODES * DINP);
  // bf16 activations
  __bf16* msgs    = allocb((size_t)BB * EDGES * HH);
  __bf16* m1      = allocb((size_t)BB * EDGES * HH);
  __bf16* xdec    = allocb((size_t)NTOK * HH);
  __bf16* xbuf    = allocb((size_t)NTOK * HH);
  __bf16* qkvbuf  = allocb((size_t)NTOK * 3 * HH);
  __bf16* attnbuf = allocb((size_t)NTOK * HH);
  __bf16* projbuf = allocb((size_t)NTOK * HH);
  __bf16* ffbuf   = allocb((size_t)NTOK * FFND);
  __bf16* membuf  = allocb((size_t)NTOK * HH);
  __bf16* hidbuf  = allocb((size_t)NTOK * HH);
  __bf16* insb    = allocb((size_t)NTOK * 32);
  __bf16* kvbuf   = qkvbuf + (size_t)NTOK * HH;      // cross-attn K|V region
  // bf16 transposed weights (Np = ceil32(Nout))
  __bf16* fc1t[3];  for (int i = 0; i < 3; ++i) fc1t[i] = allocb((size_t)2 * HH * HH);
  __bf16* fc2t[3];  for (int i = 0; i < 3; ++i) fc2t[i] = allocb((size_t)HH * HH);
  __bf16* inwt   = allocb((size_t)32 * HH);
  __bf16* o1t    = allocb((size_t)HH * HH);
  __bf16* o2t    = allocb((size_t)HH * HH);
  __bf16* o3t    = allocb((size_t)HH * 32);
  __bf16* eqkvt[2]; for (int i = 0; i < 2; ++i) eqkvt[i] = allocb((size_t)HH * 3 * HH);
  __bf16* eot[2];   for (int i = 0; i < 2; ++i) eot[i]   = allocb((size_t)HH * HH);
  __bf16* eff1t[2]; for (int i = 0; i < 2; ++i) eff1t[i] = allocb((size_t)HH * FFND);
  __bf16* eff2t[2]; for (int i = 0; i < 2; ++i) eff2t[i] = allocb((size_t)FFND * HH);
  __bf16* dqkvt[2]; for (int i = 0; i < 2; ++i) dqkvt[i] = allocb((size_t)HH * 3 * HH);
  __bf16* dot[2];   for (int i = 0; i < 2; ++i) dot[i]   = allocb((size_t)HH * HH);
  __bf16* dxqt[2];  for (int i = 0; i < 2; ++i) dxqt[i]  = allocb((size_t)HH * HH);
  __bf16* dxkvt[2]; for (int i = 0; i < 2; ++i) dxkvt[i] = allocb((size_t)HH * 2 * HH);
  __bf16* dxot[2];  for (int i = 0; i < 2; ++i) dxot[i]  = allocb((size_t)HH * HH);
  __bf16* dff1t[2]; for (int i = 0; i < 2; ++i) dff1t[i] = allocb((size_t)HH * FFND);
  __bf16* dff2t[2]; for (int i = 0; i < 2; ++i) dff2t[i] = allocb((size_t)FFND * HH);

  // ---- one-time per launch: convert+transpose weights to bf16 ----
  for (int i = 0; i < 3; ++i) {
    wtrans(stream, fc1_w + (size_t)(i + 1) * HH * 2 * HH, fc1t[i], HH, 2 * HH, 2 * HH);
    wtrans(stream, fc2_w + (size_t)(i + 1) * HH * HH,     fc2t[i], HH, HH, HH);
  }
  wtrans(stream, in_w, inwt, HH, DINP, 32);
  wtrans(stream, o1_w, o1t, HH, HH, HH);
  wtrans(stream, o2_w, o2t, HH, HH, HH);
  wtrans(stream, o3_w, o3t, DINP, HH, HH);
  for (int l = 0; l < LLAYERS; ++l) {
    wtrans(stream, e_qkv_w + (size_t)l * 3 * HH * HH, eqkvt[l], 3 * HH, HH, HH);
    wtrans(stream, e_o_w   + (size_t)l * HH * HH,     eot[l],   HH, HH, HH);
    wtrans(stream, e_ff1_w + (size_t)l * FFND * HH,   eff1t[l], FFND, HH, HH);
    wtrans(stream, e_ff2_w + (size_t)l * HH * FFND,   eff2t[l], HH, FFND, FFND);
    wtrans(stream, d_qkv_w + (size_t)l * 3 * HH * HH, dqkvt[l], 3 * HH, HH, HH);
    wtrans(stream, d_o_w   + (size_t)l * HH * HH,     dot[l],   HH, HH, HH);
    wtrans(stream, d_xqkv_w + (size_t)l * 3 * HH * HH,                  dxqt[l],  HH, HH, HH);
    wtrans(stream, d_xqkv_w + (size_t)l * 3 * HH * HH + (size_t)HH * HH, dxkvt[l], 2 * HH, HH, HH);
    wtrans(stream, d_xo_w  + (size_t)l * HH * HH,     dxot[l],  HH, HH, HH);
    wtrans(stream, d_ff1_w + (size_t)l * FFND * HH,   dff1t[l], FFND, HH, HH);
    wtrans(stream, d_ff2_w + (size_t)l * HH * FFND,   dff2t[l], HH, FFND, FFND);
  }

  // ---- init state ----
  zero_bf_kernel<<<(NTOK * HH + 255) / 256, 256, 0, stream>>>(hidbuf, NTOK * HH);
  init_ins_kernel<<<(BB * NNODES * DINP + 255) / 256, 256, 0, stream>>>(inputs, ins0);
  float* ins_cur = ins0;
  float* ins_nxt = ins1;

  for (int t = 0; t < TT; ++t) {
    // ---- message passing (gather fused into GEMM A-loads) ----
    for (int et = 1; et < ETYPES; ++et) {
      gemm(stream, A_GATHER, E_TANH, nullptr, hidbuf, fc1t[et - 1],
           fc1_b + (size_t)et * HH, nullptr, m1, BB * EDGES, HH, 2 * HH);
      gemm(stream, A_NORMAL, E_TANHEDGE, m1, nullptr, fc2t[et - 1],
           fc2_b + (size_t)et * HH, sedges, msgs, BB * EDGES, HH, HH, et, (et == 1));
    }
    agg_kernel<<<(BB * NNODES * HH + 255) / 256, 256, 0, stream>>>(msgs, xdec);

    // ---- input projection ----
    pack_ins_kernel<<<(NTOK * 32 + 255) / 256, 256, 0, stream>>>(ins_cur, insb);
    gemm(stream, A_NORMAL, E_NONE, insb, nullptr, inwt, in_b, nullptr,
         xbuf, NTOK, HH, 32);

    // ---- encoder ----
    for (int l = 0; l < LLAYERS; ++l) {
      gemm(stream, A_NORMAL, E_NONE, xbuf, nullptr, eqkvt[l],
           e_qkv_b + (size_t)l * 3 * HH, nullptr, qkvbuf, NTOK, 3 * HH, HH);
      attn_kernel<<<BB * NHEADS, 32, 0, stream>>>(qkvbuf, 3 * HH,
          qkvbuf + HH, 3 * HH, qkvbuf + 2 * HH, 3 * HH, attnbuf);
      gemm(stream, A_NORMAL, E_NONE, attnbuf, nullptr, eot[l],
           e_o_b + (size_t)l * HH, nullptr, projbuf, NTOK, HH, HH);
      ln_kernel<<<NTOK, 256, 0, stream>>>(projbuf, xbuf,
          e_ln1_g + (size_t)l * HH, e_ln1_b + (size_t)l * HH, xbuf);
      gemm(stream, A_NORMAL, E_RELU, xbuf, nullptr, eff1t[l],
           e_ff1_b + (size_t)l * FFND, nullptr, ffbuf, NTOK, FFND, HH);
      gemm(stream, A_NORMAL, E_NONE, ffbuf, nullptr, eff2t[l],
           e_ff2_b + (size_t)l * HH, nullptr, projbuf, NTOK, HH, FFND);
      ln_kernel<<<NTOK, 256, 0, stream>>>(projbuf, xbuf,
          e_ln2_g + (size_t)l * HH, e_ln2_b + (size_t)l * HH, xbuf);
    }
    ln_kernel<<<NTOK, 256, 0, stream>>>(xbuf, nullptr, e_nrm_g, e_nrm_b, membuf);

    // ---- decoder (input = aggregated messages in xdec) ----
    for (int l = 0; l < LLAYERS; ++l) {
      gemm(stream, A_NORMAL, E_NONE, xdec, nullptr, dqkvt[l],
           d_qkv_b + (size_t)l * 3 * HH, nullptr, qkvbuf, NTOK, 3 * HH, HH);
      attn_kernel<<<BB * NHEADS, 32, 0, stream>>>(qkvbuf, 3 * HH,
          qkvbuf + HH, 3 * HH, qkvbuf + 2 * HH, 3 * HH, attnbuf);
      gemm(stream, A_NORMAL, E_NONE, attnbuf, nullptr, dot[l],
           d_o_b + (size_t)l * HH, nullptr, projbuf, NTOK, HH, HH);
      ln_kernel<<<NTOK, 256, 0, stream>>>(projbuf, xdec,
          d_ln1_g + (size_t)l * HH, d_ln1_b + (size_t)l * HH, xdec);
      // cross-attention
      gemm(stream, A_NORMAL, E_NONE, xdec, nullptr, dxqt[l],
           d_xqkv_b + (size_t)l * 3 * HH, nullptr, qkvbuf, NTOK, HH, HH);
      gemm(stream, A_NORMAL, E_NONE, membuf, nullptr, dxkvt[l],
           d_xqkv_b + (size_t)l * 3 * HH + HH, nullptr, kvbuf, NTOK, 2 * HH, HH);
      attn_kernel<<<BB * NHEADS, 32, 0, stream>>>(qkvbuf, HH,
          kvbuf, 2 * HH, kvbuf + HH, 2 * HH, attnbuf);
      gemm(stream, A_NORMAL, E_NONE, attnbuf, nullptr, dxot[l],
           d_xo_b + (size_t)l * HH, nullptr, projbuf, NTOK, HH, HH);
      ln_kernel<<<NTOK, 256, 0, stream>>>(projbuf, xdec,
          d_ln2_g + (size_t)l * HH, d_ln2_b + (size_t)l * HH, xdec);
      // FFN
      gemm(stream, A_NORMAL, E_RELU, xdec, nullptr, dff1t[l],
           d_ff1_b + (size_t)l * FFND, nullptr, ffbuf, NTOK, FFND, HH);
      gemm(stream, A_NORMAL, E_NONE, ffbuf, nullptr, dff2t[l],
           d_ff2_b + (size_t)l * HH, nullptr, projbuf, NTOK, HH, FFND);
      ln_kernel<<<NTOK, 256, 0, stream>>>(projbuf, xdec,
          d_ln3_g + (size_t)l * HH, d_ln3_b + (size_t)l * HH, xdec);
    }
    ln_kernel<<<NTOK, 256, 0, stream>>>(xdec, nullptr, d_nrm_g, d_nrm_b, hidbuf);

    // ---- output head: pred = ins + o3(relu(o2(relu(o1(hid))))) ----
    gemm(stream, A_NORMAL, E_RELU, hidbuf, nullptr, o1t, o1_b, nullptr,
         attnbuf, NTOK, HH, HH);
    gemm(stream, A_NORMAL, E_RELU, attnbuf, nullptr, o2t, o2_b, nullptr,
         projbuf, NTOK, HH, HH);
    gemm(stream, A_NORMAL, E_ADDRES, projbuf, nullptr, o3t, o3_b, ins_cur,
         ins_nxt, NTOK, DINP, HH);
    write_out_kernel<<<(BB * NNODES * DINP + 255) / 256, 256, 0, stream>>>(ins_nxt, out, t);

    float* tmp = ins_cur; ins_cur = ins_nxt; ins_nxt = tmp;   // pred -> next ins
  }
}
